// MCNET_28991029248223
// MI455X (gfx1250) — 8-bit matrix operand order repaired and checked by reading, compiled, not run on hardware
//
#include <hip/hip_runtime.h>
#include <stdint.h>

typedef __attribute__((ext_vector_type(8))) int v8i;

__global__ __launch_bounds__(256) void bnn_conv1_pool_sign(
    const float* __restrict__ x,
    const float* __restrict__ w1,
    int8_t* __restrict__ out)
{
    __shared__ float sw[8 * 3 * 3 * 3];
    const int tid = threadIdx.x;
    if (tid < 216) {
        float wv = w1[tid];
        sw[tid] = (wv > 0.f) ? 1.f : ((wv < 0.f) ? -1.f : 0.f);
    }
    __syncthreads();

    const int gid = blockIdx.x * 256 + tid;
    const int total = 32 * 256 * 256;
    if (gid >= total) return;
    const int px = gid & 255;
    const int py = (gid >> 8) & 255;
    const int n  = gid >> 16;

    float p[3][4][4];
#pragma unroll
    for (int c = 0; c < 3; ++c) {
#pragma unroll
        for (int r = 0; r < 4; ++r) {
            const int iy = 2 * py - 1 + r;
#pragma unroll
            for (int cc = 0; cc < 4; ++cc) {
                const int ix = 2 * px - 1 + cc;
                float v = 1.0f;
                if (iy >= 0 && iy < 512 && ix >= 0 && ix < 512)
                    v = x[(((size_t)n * 3 + c) * 512 + iy) * 512 + ix];
                p[c][r][cc] = v;
            }
        }
    }

    int8_t* op = out + ((((size_t)n * 256 + py) * 256 + px) * 8);
#pragma unroll
    for (int co = 0; co < 8; ++co) {
        float m = -2.0f;
#pragma unroll
        for (int sy = 0; sy < 2; ++sy) {
#pragma unroll
            for (int sx = 0; sx < 2; ++sx) {
                float s = 0.f;
#pragma unroll
                for (int c = 0; c < 3; ++c)
#pragma unroll
                    for (int ty = 0; ty < 3; ++ty)
#pragma unroll
                        for (int tx = 0; tx < 3; ++tx)
                            s += p[c][sy + ty][sx + tx] * sw[((co * 3 + c) * 3 + ty) * 3 + tx];
                s = fminf(1.f, fmaxf(-1.f, s));
                m = fmaxf(m, s);
            }
        }
        op[co] = (m > 0.f) ? (int8_t)1 : ((m < 0.f) ? (int8_t)-1 : (int8_t)0);
    }
}

template <int CIN, int COUT, int HIN, int WIN, bool FINAL>
__global__ __launch_bounds__(256) void bnn_ternary_conv_wmma(
    const int8_t* __restrict__ in,
    const float* __restrict__ w,
    int8_t* __restrict__ out8,
    float* __restrict__ outf)
{
    constexpr int HOUT = HIN - 2, WOUT = WIN - 2;
    constexpr int KV = 9 * CIN;
    constexpr int CHUNKS = (KV + 63) / 64;
    constexpr int NT = (COUT + 15) / 16;
    constexpr int TW = 18;
    constexpr int NDW = 10 * TW * CIN / 4;

    __shared__ alignas(16) int8_t sact[10 * TW * CIN];
    __shared__ alignas(4)  int8_t swt[COUT * KV];

    const int tid = threadIdx.x;
    const int n  = blockIdx.z;
    const int y0 = blockIdx.y * 8;
    const int x0 = blockIdx.x * 16;

    const int8_t* gbase = in + (size_t)n * HIN * WIN * CIN;
    for (int idx = tid; idx < NDW; idx += 256) {
        const int b  = idx * 4;
        const int rr = b / (TW * CIN);
        const int r2 = b % (TW * CIN);
        const int cc = r2 / CIN;
        const int ch = r2 % CIN;
        const int iy = y0 + rr, ix = x0 + cc;
        if (iy < HIN && ix < WIN) {
            const uint32_t lds_off = (uint32_t)(size_t)&sact[b];
            const uint64_t gaddr =
                (uint64_t)(size_t)(gbase + ((size_t)iy * WIN + ix) * CIN + ch);
            asm volatile("global_load_async_to_lds_b32 %0, %1, off"
                         :: "v"(lds_off), "v"(gaddr) : "memory");
        } else {
            *(uint32_t*)&sact[b] = 0u;
        }
    }
    for (int idx = tid; idx < COUT * KV; idx += 256) {
        const int co = idx / KV;
        const int k  = idx % KV;
        const int tap = k / CIN, ci = k % CIN;
        const int ty = tap / 3, tx = tap % 3;
        const float wv = w[((co * CIN + ci) * 3 + ty) * 3 + tx];
        swt[idx] = (wv > 0.f) ? (int8_t)1 : ((wv < 0.f) ? (int8_t)-1 : (int8_t)0);
    }
    asm volatile("s_wait_asynccnt 0" ::: "memory");
    __syncthreads();

    const int wave = tid >> 5;
    const int lane = tid & 31;
    const int half = (lane >> 4) & 1;
    const int y = y0 + wave;
    if (y >= HOUT) return;

    const int m = lane & 15;
    v8i Afr[CHUNKS];
#pragma unroll
    for (int c = 0; c < CHUNKS; ++c) {
#pragma unroll
        for (int v = 0; v < 8; ++v) {
            const int kb = c * 64 + ((v >> 1) * 4 + (v & 1) + half * 2) * 4;
            uint32_t d = 0;
            if (kb < KV) {
                const int ty  = kb / (3 * CIN);
                const int rem = kb % (3 * CIN);
                d = *(const uint32_t*)&sact[((wave + ty) * TW + m) * CIN + rem];
            }
            Afr[c][v] = (int)d;
        }
    }

#pragma unroll
    for (int t = 0; t < NT; ++t) {
        const int ncol = t * 16 + (lane & 15);

        v8i Bfr[CHUNKS];
#pragma unroll
        for (int c = 0; c < CHUNKS; ++c) {
#pragma unroll
            for (int v = 0; v < 8; ++v) {
                const int k = c * 64 + ((v >> 1) * 4 + (v & 1) + half * 2) * 4;
                uint32_t d = 0;
                if (k < KV && ncol < COUT)
                    d = *(const uint32_t*)&swt[ncol * KV + k];
                Bfr[c][v] = (int)d;
            }
        }

        v8i acc = {};
#pragma unroll
        for (int c = 0; c < CHUNKS; ++c) {
            acc = __builtin_amdgcn_wmma_i32_16x16x64_iu8(
                true, Afr[c], true, Bfr[c], acc, false, false);
        }

        if (ncol < COUT) {
#pragma unroll
            for (int r = 0; r < 8; ++r) {
                const int xm = x0 + r + half * 8;
                if (xm < WOUT) {
                    int v = acc[r];
                    v = v > 1 ? 1 : (v < -1 ? -1 : v);
                    if (FINAL) {
                        outf[(((size_t)n * COUT + ncol) * HOUT + y) * WOUT + xm] = (float)v;
                    } else {
                        out8[(((size_t)n * HOUT + y) * WOUT + xm) * COUT + ncol] = (int8_t)v;
                    }
                }
            }
        }
    }
}

extern "C" void kernel_launch(void* const* d_in, const int* in_sizes, int n_in,
                              void* d_out, int out_size, void* d_ws, size_t ws_size,
                              hipStream_t stream) {
    (void)in_sizes; (void)n_in; (void)out_size; (void)ws_size;
    const float* x  = (const float*)d_in[0];
    const float* w1 = (const float*)d_in[1];
    const float* w2 = (const float*)d_in[2];
    const float* w3 = (const float*)d_in[3];
    const float* w4 = (const float*)d_in[4];
    float* out = (float*)d_out;
    int8_t* ws = (int8_t*)d_ws;

    const size_t buf2_off = 0;
    const size_t buf3_off = (size_t)32 * 254 * 254 * 16;
    const size_t buf1_off = buf3_off;

    bnn_conv1_pool_sign<<<(32 * 256 * 256) / 256, 256, 0, stream>>>(
        x, w1, ws + buf1_off);

    {
        dim3 g((254 + 15) / 16, (254 + 7) / 8, 32);
        bnn_ternary_conv_wmma<8, 16, 256, 256, false><<<g, 256, 0, stream>>>(
            ws + buf1_off, w2, ws + buf2_off, nullptr);
    }
    {
        dim3 g((252 + 15) / 16, (252 + 7) / 8, 32);
        bnn_ternary_conv_wmma<16, 32, 254, 254, false><<<g, 256, 0, stream>>>(
            ws + buf2_off, w3, ws + buf3_off, nullptr);
    }
    {
        dim3 g((250 + 15) / 16, (250 + 7) / 8, 32);
        bnn_ternary_conv_wmma<32, 2, 252, 252, true><<<g, 256, 0, stream>>>(
            ws + buf3_off, w4, nullptr, out);
    }
}
